// RoPEAttention_21663815041590
// MI455X (gfx1250) — matrix operand order repaired and checked by reading, not compiled, not run on hardware
//
#include <hip/hip_runtime.h>
#include <stdint.h>

typedef __attribute__((ext_vector_type(16))) __bf16 v16bf;
typedef __attribute__((ext_vector_type(8)))  float  v8f;

#define F_DIM 512
#define N_SEQ 4096
#define HEADS 8
#define HDIM  64

__device__ __forceinline__ unsigned short f32_to_bf16(float f) {
  union { float f; uint32_t u; } v; v.f = f;
  uint32_t r = v.u + 0x7FFFu + ((v.u >> 16) & 1u);   // round-to-nearest-even
  return (unsigned short)(r >> 16);
}

__device__ __forceinline__ v8f zero8() {
  v8f z = {0.f, 0.f, 0.f, 0.f, 0.f, 0.f, 0.f, 0.f};
  return z;
}

union FragU { v16bf v; uint32_t u[8]; };

// A-matrix fragment (16x32 bf16, M x K), ISA 7.12.2 layout:
// lanes 0-15: M=lane, K pairs {0..7,16..23}; lanes 16-31: M=lane-16, K pairs +8.
__device__ __forceinline__ v16bf load_a_frag(const unsigned short* s, int stride, int lane) {
  FragU r;
  const unsigned short* row = s + (lane & 15) * stride;
  const int kg = (lane >> 4) * 8;
#pragma unroll
  for (int j = 0; j < 8; ++j) {
    const int kb = ((j < 4) ? (2 * j) : (16 + 2 * (j - 4))) + kg;
    r.u[j] = *(const uint32_t*)(row + kb);
  }
  return r.v;
}

// B-matrix fragment (32x16 bf16, K x N) from an LDS tile stored [N][K]:
// lanes 0-15: N=lane, K pairs {0..7,16..23}; lanes 16-31: N=lane-16, K pairs +8. kextra shifts K base.
__device__ __forceinline__ v16bf load_b_frag(const unsigned short* s, int stride, int lane, int kextra) {
  FragU r;
  const unsigned short* row = s + (lane & 15) * stride;
  const int koff = ((lane >> 4) * 8) + kextra;
#pragma unroll
  for (int j = 0; j < 8; ++j)
    r.u[j] = *(const uint32_t*)(row + koff + ((j < 4) ? (2 * j) : (16 + 2 * (j - 4))));
  return r.v;
}

__device__ __forceinline__ v8f wmma_bf16(v16bf a, v16bf b, v8f c) {
  return __builtin_amdgcn_wmma_f32_16x16x32_bf16(false, a, false, b, (short)0, c, false, false);
}

// Async global->LDS copy, 16 bytes per lane (GVS addressing: saddr + 32-bit voffset).
// Tracked by ASYNCcnt; in-order completion lets us fence with a partial wait.
__device__ __forceinline__ void async_b128(uint32_t lds_byte_addr, const void* base,
                                           uint32_t byte_off) {
  asm volatile("global_load_async_to_lds_b128 %0, %1, %2 offset:0"
               :: "v"(lds_byte_addr), "v"(byte_off), "s"(base)
               : "memory");
}

// ---------------------------------------------------------------------------
// GEMM: Y[n,f] = sum_c X[n,c] * W[f,c] + bias[f]
// OUT_MODE 0: f32 flat [N][F] (final output projection)
// OUT_MODE 1: bf16 head-split [H][N][64] with axial-2D RoPE (Q/K)
// OUT_MODE 2: bf16 head-split TRANSPOSED [H][64][N] (V, pre-transposed for PV B-tiles)
// ---------------------------------------------------------------------------
template <bool IN_BF16, int OUT_MODE>
__global__ void __launch_bounds__(256) gemm_wmma_kernel(
    const void* __restrict__ Xv, const float* __restrict__ W,
    const float* __restrict__ bias, void* __restrict__ Out)
{
  __shared__ alignas(16) unsigned short As[128][36];  // [BM][BK+4]
  __shared__ alignas(16) unsigned short Bs[64][36];   // [BN][BK+4]

  const int tid   = threadIdx.x;
  const int lane  = tid & 31;
  const int wave  = tid >> 5;
  const int waveM = wave >> 1;
  const int waveN = wave & 1;
  const int blockM = blockIdx.y * 128;
  const int blockN = blockIdx.x * 64;

  v8f acc[2][2];
#pragma unroll
  for (int i = 0; i < 2; ++i)
#pragma unroll
    for (int j = 0; j < 2; ++j) acc[i][j] = zero8();

  for (int k0 = 0; k0 < F_DIM; k0 += 32) {
    if (IN_BF16) {
      const unsigned short* X = (const unsigned short*)Xv;
#pragma unroll
      for (int it = 0; it < 8; ++it) {            // 2048 dwords = 128x32 bf16
        int i = tid + it * 256;
        int r = i >> 4, c2 = i & 15;
        uint32_t d = *(const uint32_t*)(X + (size_t)(blockM + r) * F_DIM + k0 + c2 * 2);
        *(uint32_t*)&As[r][c2 * 2] = d;
      }
    } else {
      const float* X = (const float*)Xv;
#pragma unroll
      for (int it = 0; it < 16; ++it) {           // 4096 f32 -> bf16
        int i = tid + it * 256;
        int r = i >> 5, c = i & 31;
        As[r][c] = f32_to_bf16(X[(size_t)(blockM + r) * F_DIM + k0 + c]);
      }
    }
#pragma unroll
    for (int it = 0; it < 8; ++it) {              // 2048 f32 -> bf16 (weights)
      int i = tid + it * 256;
      int n = i >> 5, c = i & 31;
      Bs[n][c] = f32_to_bf16(W[(size_t)(blockN + n) * F_DIM + k0 + c]);
    }
    __syncthreads();

#pragma unroll
    for (int tm = 0; tm < 2; ++tm) {
      v16bf a = load_a_frag(&As[waveM * 32 + tm * 16][0], 36, lane);
#pragma unroll
      for (int tn = 0; tn < 2; ++tn) {
        v16bf b = load_b_frag(&Bs[waveN * 32 + tn * 16][0], 36, lane, 0);
        acc[tm][tn] = wmma_bf16(a, b, acc[tm][tn]);
      }
    }
    __syncthreads();
  }

  const int halfSel = (lane >> 4) * 8;
  const int colLane = lane & 15;
#pragma unroll
  for (int tm = 0; tm < 2; ++tm) {
#pragma unroll
    for (int tn = 0; tn < 2; ++tn) {
#pragma unroll
      for (int r = 0; r < 8; ++r) {
        int row = blockM + waveM * 32 + tm * 16 + halfSel + r;
        int col = blockN + waveN * 32 + tn * 16 + colLane;
        float y = acc[tm][tn][r] + bias[col];
        if (OUT_MODE == 0) {
          ((float*)Out)[(size_t)row * F_DIM + col] = y;
        } else {
          float val = y;
          if (OUT_MODE == 1) {
            // RoPE: feature pairs are adjacent cols == adjacent lanes.
            float partner = __shfl_xor(y, 1, 32);
            int d = col & (HDIM - 1);
            int p = d >> 1;
            float pos = (p < 16) ? (float)(row & 63) : (float)(row >> 6); // x_m / y_m
            int fidx = p & 15;
            float ang = pos * __expf(-(float)fidx * 0.5756462732485115f); // theta^(-f/16)
            float cs = __cosf(ang), sn = __sinf(ang);
            bool isX = ((d & 1) == 0);
            float xv = isX ? y : partner;
            float yv = isX ? partner : y;
            val = isX ? (xv * cs - yv * sn) : (xv * sn + yv * cs);
          }
          int hh = col >> 6, d2 = col & (HDIM - 1);
          if (OUT_MODE == 2)   // V: [H][64][N] pre-transposed
            ((unsigned short*)Out)[((size_t)hh * HDIM + d2) * N_SEQ + row] = f32_to_bf16(val);
          else                 // Q/K: [H][N][64]
            ((unsigned short*)Out)[((size_t)hh * N_SEQ + row) * HDIM + d2] = f32_to_bf16(val);
        }
      }
    }
  }
}

// ---------------------------------------------------------------------------
// Flash attention: one (head, 128-query) tile per block; 8 waves x 16 rows.
// KV chunk = 64. K [kv][d] and V [d][kv] tiles double-buffered in LDS and
// filled with global_load_async_to_lds_b128 (ASYNCcnt-fenced).
// ---------------------------------------------------------------------------
__global__ void __launch_bounds__(256) flash_attn_kernel(
    const unsigned short* __restrict__ qh, const unsigned short* __restrict__ kh,
    const unsigned short* __restrict__ vhT, unsigned short* __restrict__ outp)
{
  __shared__ alignas(16) unsigned short Kt[2][64][72];  // [buf][kv][d]  (144B rows, 16B-aligned chunks)
  __shared__ alignas(16) unsigned short Vt[2][64][72];  // [buf][d][kv]
  __shared__ alignas(16) unsigned short Pb[8][16][68];  // per-wave P transpose buffer

  const int head  = blockIdx.y;
  const int tid   = threadIdx.x;
  const int lane  = tid & 31;
  const int wave  = tid >> 5;
  const int qBase = blockIdx.x * 128 + wave * 16;

  const unsigned short* Kg = kh  + (size_t)head * N_SEQ * HDIM;   // [kv][d]
  const unsigned short* Vg = vhT + (size_t)head * HDIM * N_SEQ;   // [d][kv]

  // Per-thread staging: 512 x 16B chunks per tile; thread covers chunks tid, tid+256.
  const int r0 = tid >> 3;            // row 0..31
  const int c0 = tid & 7;             // 16B chunk within 128B row payload
  const int r1 = r0 + 32;             // row 32..63

  auto prefetch = [&](int buf, int kv0) {
    async_b128((uint32_t)(uintptr_t)&Kt[buf][r0][c0 * 8], Kg,
               (uint32_t)((((kv0 + r0) * HDIM) + c0 * 8) * 2));
    async_b128((uint32_t)(uintptr_t)&Kt[buf][r1][c0 * 8], Kg,
               (uint32_t)((((kv0 + r1) * HDIM) + c0 * 8) * 2));
    async_b128((uint32_t)(uintptr_t)&Vt[buf][r0][c0 * 8], Vg,
               (uint32_t)(((r0 * N_SEQ) + kv0 + c0 * 8) * 2));
    async_b128((uint32_t)(uintptr_t)&Vt[buf][r1][c0 * 8], Vg,
               (uint32_t)(((r1 * N_SEQ) + kv0 + c0 * 8) * 2));
  };

  prefetch(0, 0);   // kick off first tile before doing anything else

  const unsigned short* Q = qh + ((size_t)head * N_SEQ + qBase) * HDIM;
  v16bf qf[2];
#pragma unroll
  for (int ks = 0; ks < 2; ++ks) qf[ks] = load_a_frag(Q + ks * 32, HDIM, lane);

  v8f Ov[4];
#pragma unroll
  for (int nt = 0; nt < 4; ++nt) Ov[nt] = zero8();
  float mrow[8], lrow[8];
#pragma unroll
  for (int r = 0; r < 8; ++r) { mrow[r] = -3.0e38f; lrow[r] = 0.f; }

  const int NCHUNK = N_SEQ / 64;
  for (int ic = 0; ic < NCHUNK; ++ic) {
    const int cur = ic & 1;
    if (ic + 1 < NCHUNK) {
      prefetch(cur ^ 1, (ic + 1) * 64);                    // overlap next tile
      asm volatile("s_wait_asynccnt 4" ::: "memory");      // current tile's 4 copies done
    } else {
      asm volatile("s_wait_asynccnt 0" ::: "memory");
    }
    __syncthreads();

    // S = Q * K^T : 16x64 scores per wave
    v8f S[4];
#pragma unroll
    for (int nt = 0; nt < 4; ++nt) S[nt] = zero8();
#pragma unroll
    for (int ks = 0; ks < 2; ++ks) {
#pragma unroll
      for (int nt = 0; nt < 4; ++nt) {
        v16bf b = load_b_frag(&Kt[cur][nt * 16][0], 72, lane, ks * 32);
        S[nt] = wmma_bf16(qf[ks], b, S[nt]);
      }
    }

    // Online softmax (scale = 1/sqrt(64)); row reduction over 16-lane half.
    const float scale = 0.125f;
#pragma unroll
    for (int r = 0; r < 8; ++r) {
      float s0 = S[0][r] * scale, s1 = S[1][r] * scale;
      float s2 = S[2][r] * scale, s3 = S[3][r] * scale;
      float cm = fmaxf(fmaxf(s0, s1), fmaxf(s2, s3));
      cm = fmaxf(cm, __shfl_xor(cm, 1, 32));
      cm = fmaxf(cm, __shfl_xor(cm, 2, 32));
      cm = fmaxf(cm, __shfl_xor(cm, 4, 32));
      cm = fmaxf(cm, __shfl_xor(cm, 8, 32));
      float mnew  = fmaxf(mrow[r], cm);
      float alpha = __expf(mrow[r] - mnew);
      mrow[r] = mnew;
      float p0 = __expf(s0 - mnew), p1 = __expf(s1 - mnew);
      float p2 = __expf(s2 - mnew), p3 = __expf(s3 - mnew);
      S[0][r] = p0; S[1][r] = p1; S[2][r] = p2; S[3][r] = p3;
      float rs = p0 + p1 + p2 + p3;
      rs += __shfl_xor(rs, 1, 32);
      rs += __shfl_xor(rs, 2, 32);
      rs += __shfl_xor(rs, 4, 32);
      rs += __shfl_xor(rs, 8, 32);
      lrow[r] = lrow[r] * alpha + rs;
#pragma unroll
      for (int nt = 0; nt < 4; ++nt) Ov[nt][r] *= alpha;
    }

    // Transpose P (C-layout -> A-layout) via per-wave LDS buffer.
    const int prow = (lane >> 4) * 8;
    const int pcol = lane & 15;
#pragma unroll
    for (int nt = 0; nt < 4; ++nt)
#pragma unroll
      for (int r = 0; r < 8; ++r)
        Pb[wave][prow + r][nt * 16 + pcol] = f32_to_bf16(S[nt][r]);

    // O += P * V   (V tile already [d][kv] -> direct B-fragment loads)
#pragma unroll
    for (int ks = 0; ks < 2; ++ks) {
      v16bf pa = load_a_frag(&Pb[wave][0][0] + ks * 32, 68, lane);
#pragma unroll
      for (int nt = 0; nt < 4; ++nt) {
        v16bf vb = load_b_frag(&Vt[cur][nt * 16][0], 72, lane, ks * 32);
        Ov[nt] = wmma_bf16(pa, vb, Ov[nt]);
      }
    }
    __syncthreads();
  }

  // Normalize and write merged-head bf16 [N][F] for the output projection.
  const int orow = (lane >> 4) * 8;
  const int ocol = lane & 15;
#pragma unroll
  for (int nt = 0; nt < 4; ++nt) {
#pragma unroll
    for (int r = 0; r < 8; ++r) {
      int row = qBase + orow + r;
      int d = nt * 16 + ocol;
      float o = Ov[nt][r] / lrow[r];
      outp[(size_t)row * F_DIM + head * HDIM + d] = f32_to_bf16(o);
    }
  }
}

// ---------------------------------------------------------------------------
extern "C" void kernel_launch(void* const* d_in, const int* in_sizes, int n_in,
                              void* d_out, int out_size, void* d_ws, size_t ws_size,
                              hipStream_t stream) {
  (void)in_sizes; (void)n_in; (void)out_size; (void)ws_size;
  const float* q  = (const float*)d_in[0];
  const float* k  = (const float*)d_in[1];
  const float* v  = (const float*)d_in[2];
  const float* Wq = (const float*)d_in[3];
  const float* bq = (const float*)d_in[4];
  const float* Wk = (const float*)d_in[5];
  const float* bk = (const float*)d_in[6];
  const float* Wv = (const float*)d_in[7];
  const float* bv = (const float*)d_in[8];
  const float* Wo = (const float*)d_in[9];
  const float* bo = (const float*)d_in[10];
  // d_in[11], d_in[12] = h, w (hardcoded 64x64; N_SEQ = 4096 = 64*64)

  unsigned short* qhb = (unsigned short*)d_ws;                       // [H][N][64] bf16
  unsigned short* khb = qhb + (size_t)HEADS * N_SEQ * HDIM;          // [H][N][64]
  unsigned short* vhb = khb + (size_t)HEADS * N_SEQ * HDIM;          // [H][64][N] (transposed)
  unsigned short* att = vhb + (size_t)HEADS * N_SEQ * HDIM;          // [N][F]

  dim3 blk(256);
  dim3 gGemm(F_DIM / 64, N_SEQ / 128);
  gemm_wmma_kernel<false, 1><<<gGemm, blk, 0, stream>>>(q, Wq, bq, qhb);
  gemm_wmma_kernel<false, 1><<<gGemm, blk, 0, stream>>>(k, Wk, bk, khb);
  gemm_wmma_kernel<false, 2><<<gGemm, blk, 0, stream>>>(v, Wv, bv, vhb);
  flash_attn_kernel<<<dim3(N_SEQ / 128, HEADS), blk, 0, stream>>>(qhb, khb, vhb, att);
  gemm_wmma_kernel<true, 0><<<gGemm, blk, 0, stream>>>(att, Wo, bo, (float*)d_out);
}
